// Attention_24455543784003
// MI455X (gfx1250) — matrix operand order repaired and checked by reading, not compiled, not run on hardware
//
#include <hip/hip_runtime.h>
#include <hip/hip_bf16.h>

#define B_ 128
#define H_ 32
#define D_ 128
#define S_ 64
#define START_ 32
#define STEPS_ 32
#define TB 16
#define SCALE_ 0.125f

typedef __attribute__((ext_vector_type(16))) __bf16 v16bf;
typedef __attribute__((ext_vector_type(8)))  __bf16 v8bf;
typedef __attribute__((ext_vector_type(8)))  float  v8f;

// ---------------------------------------------------------------------------
// Kernel 1: convert fp32 weights to bf16, transposed per head:
//   Wt[mat][h][e][d] = W_mat[h][d][e]
// so that WMMA B-fragments (K contiguous per N) are contiguous 32B loads.
// ---------------------------------------------------------------------------
__global__ void cvt_weights(const float* __restrict__ Wq,
                            const float* __restrict__ Wk,
                            const float* __restrict__ Wv,
                            const float* __restrict__ Wo,
                            __bf16* __restrict__ out) {
    int idx = blockIdx.x * blockDim.x + threadIdx.x;   // 4*H*D*D threads
    const int per = H_ * D_ * D_;
    int mat = idx / per;
    int r   = idx % per;
    int h   = r / (D_ * D_);
    int rr  = r % (D_ * D_);
    int e   = rr / D_;
    int d   = rr % D_;
    const float* W = (mat == 0) ? Wq : (mat == 1) ? Wk : (mat == 2) ? Wv : Wo;
    out[idx] = (__bf16)W[h * D_ * D_ + d * D_ + e];
}

static __device__ inline v8f wmma_bf16(v16bf a, v16bf b, v8f c) {
    return __builtin_amdgcn_wmma_f32_16x16x32_bf16(
        /*neg_a=*/false, a, /*neg_b=*/false, b,
        /*c_mod=*/(short)0, c, /*reuse_a=*/false, /*reuse_b=*/false);
}

// ---------------------------------------------------------------------------
// Kernel 2: one workgroup = (head h, 16-batch tile). 256 threads = 8 waves.
// Runs all 32 decode steps with dynamic K/V rows resident in LDS (bf16),
// projections done with v_wmma_f32_16x16x32_bf16 (fp32 accumulate).
// ---------------------------------------------------------------------------
__global__ __launch_bounds__(256) void attn_decode(
    const float* __restrict__ x_in, const float* __restrict__ k_in,
    const float* __restrict__ v_in, const __bf16* __restrict__ Wt,
    float* __restrict__ k_out, float* __restrict__ v_out,
    float* __restrict__ x_out)
{
    extern __shared__ char smem[];
    __bf16* Xb = (__bf16*)smem;            // [TB][D]  current x (bf16)   4KB
    __bf16* Ab = Xb + TB * D_;             // [TB][D]  attn out (bf16)    4KB
    float*  Qs = (float*)(Ab + TB * D_);   // [TB][D]  q (fp32)           8KB
    float*  Sc = Qs + TB * D_;             // [TB][S]  scores/probs       4KB
    __bf16* Kd = (__bf16*)(Sc + TB * S_);  // [TB][32][D] dyn K rows    128KB
    __bf16* Vd = Kd + TB * STEPS_ * D_;    // [TB][32][D] dyn V rows    128KB

    const int t     = threadIdx.x;
    const int wave  = t >> 5;       // 0..7  -> N-tile for GEMMs
    const int lane  = t & 31;
    const int mlane = lane & 15;
    const int hlf   = lane >> 4;
    const int h     = blockIdx.x >> 3;
    const int b0    = (blockIdx.x & 7) * TB;

    const __bf16* WqT = Wt;
    const __bf16* WkT = Wt + (size_t)H_ * D_ * D_;
    const __bf16* WvT = Wt + (size_t)2 * H_ * D_ * D_;
    const __bf16* WoT = Wt + (size_t)3 * H_ * D_ * D_;

    // ---- init Xb from x input ----
    for (int i = t; i < TB * D_; i += 256) {
        int b = i >> 7, d = i & 127;
        Xb[i] = (__bf16)x_in[((size_t)(b0 + b) * H_ + h) * D_ + d];
    }
    // ---- copy static K/V rows (s < 32) straight to outputs ----
    for (int i = t; i < TB * START_ * D_ / 4; i += 256) {
        int b = i / (START_ * D_ / 4);
        int r = i % (START_ * D_ / 4);
        size_t base = ((size_t)((b0 + b) * H_ + h) * S_) * D_;
        ((float4*)(k_out + base))[r] = ((const float4*)(k_in + base))[r];
        ((float4*)(v_out + base))[r] = ((const float4*)(v_in + base))[r];
    }
    __syncthreads();

    for (int step = 0; step < STEPS_; ++step) {
        const int gen = START_ + step;

        // ---------- phase 1: Q/Knew/Vnew = Xb @ W  (WMMA) ----------
        const __bf16* Wh[3] = { WqT + (size_t)h * D_ * D_,
                                WkT + (size_t)h * D_ * D_,
                                WvT + (size_t)h * D_ * D_ };
        for (int mat = 0; mat < 3; ++mat) {
            v8f c = {};
            #pragma unroll
            for (int k0 = 0; k0 < D_; k0 += 32) {
                union { v16bf v; v8bf p[2]; } a;
                a.p[0] = *(const v8bf*)&Xb[mlane * D_ + k0 + hlf * 16];
                a.p[1] = *(const v8bf*)&Xb[mlane * D_ + k0 + hlf * 16 + 8];
                v16bf bfr = *(const v16bf*)(Wh[mat] +
                        (size_t)(wave * 16 + mlane) * D_ + k0 + hlf * 16);
                c = wmma_bf16(a.v, bfr, c);
            }
            if (mat == 0) {
                #pragma unroll
                for (int j = 0; j < 8; ++j)
                    Qs[(hlf * 8 + j) * D_ + wave * 16 + mlane] = c[j];
            } else {
                float*  gout = (mat == 1) ? k_out : v_out;
                __bf16* ldsd = (mat == 1) ? Kd : Vd;
                #pragma unroll
                for (int j = 0; j < 8; ++j) {
                    int m = hlf * 8 + j;
                    int d = wave * 16 + mlane;
                    gout[(((size_t)(b0 + m) * H_ + h) * S_ + gen) * D_ + d] = c[j];
                    ldsd[(m * STEPS_ + step) * D_ + d] = (__bf16)c[j];
                }
            }
        }
        __syncthreads();

        // ---------- phase 2: scores = SCALE * K . q ----------
        #pragma unroll
        for (int p4 = 0; p4 < 4; ++p4) {
            int pair = t * 4 + p4;          // 1024 (b,s) pairs
            int b = pair >> 6;
            int s = pair & 63;
            const float* q = &Qs[b * D_];
            float acc = 0.f;
            if (s >= START_ && s <= gen) {  // row written this run -> LDS
                const __bf16* kr = &Kd[(b * STEPS_ + (s - START_)) * D_];
                #pragma unroll 8
                for (int d = 0; d < D_; ++d) acc += (float)kr[d] * q[d];
            } else {                         // stale/original row -> input
                const float* kr = &k_in[(((size_t)(b0 + b) * H_ + h) * S_ + s) * D_];
                for (int d = 0; d < D_; d += 4) {
                    float4 kk = *(const float4*)&kr[d];
                    acc += kk.x * q[d] + kk.y * q[d + 1] +
                           kk.z * q[d + 2] + kk.w * q[d + 3];
                }
            }
            Sc[b * S_ + s] = acc * SCALE_;
        }
        __syncthreads();

        // ---------- phase 3: softmax (one 16-lane group per batch) ----------
        {
            int b = t >> 4;
            int l = t & 15;
            float vals[4], vmax = -1e30f;
            #pragma unroll
            for (int i = 0; i < 4; ++i) {
                vals[i] = Sc[b * S_ + l + i * 16];
                vmax = fmaxf(vmax, vals[i]);
            }
            for (int m = 1; m < 16; m <<= 1)
                vmax = fmaxf(vmax, __shfl_xor(vmax, m, 32));
            float sum = 0.f;
            #pragma unroll
            for (int i = 0; i < 4; ++i) { vals[i] = __expf(vals[i] - vmax); sum += vals[i]; }
            for (int m = 1; m < 16; m <<= 1) sum += __shfl_xor(sum, m, 32);
            float inv = 1.f / sum;
            #pragma unroll
            for (int i = 0; i < 4; ++i) Sc[b * S_ + l + i * 16] = vals[i] * inv;
        }
        __syncthreads();

        // ---------- phase 4: attn_out = P @ V ----------
        {
            int b = t >> 4;
            int l = t & 15;
            const int dbase = l * 8;
            float acc[8] = {0, 0, 0, 0, 0, 0, 0, 0};
            for (int s = 0; s < S_; ++s) {
                float p = Sc[b * S_ + s];
                if (s >= START_ && s <= gen) {
                    v8bf vv = *(const v8bf*)&Vd[(b * STEPS_ + (s - START_)) * D_ + dbase];
                    #pragma unroll
                    for (int j = 0; j < 8; ++j) acc[j] += p * (float)vv[j];
                } else {
                    const float4* vr = (const float4*)
                        &v_in[(((size_t)(b0 + b) * H_ + h) * S_ + s) * D_ + dbase];
                    float4 a0 = vr[0], a1 = vr[1];
                    acc[0] += p * a0.x; acc[1] += p * a0.y;
                    acc[2] += p * a0.z; acc[3] += p * a0.w;
                    acc[4] += p * a1.x; acc[5] += p * a1.y;
                    acc[6] += p * a1.z; acc[7] += p * a1.w;
                }
            }
            #pragma unroll
            for (int j = 0; j < 8; ++j) Ab[b * D_ + dbase + j] = (__bf16)acc[j];
        }
        __syncthreads();

        // ---------- phase 5: x_next = attn_out @ Wo  (WMMA) ----------
        {
            v8f c = {};
            const __bf16* Wo_h = WoT + (size_t)h * D_ * D_;
            #pragma unroll
            for (int k0 = 0; k0 < D_; k0 += 32) {
                union { v16bf v; v8bf p[2]; } a;
                a.p[0] = *(const v8bf*)&Ab[mlane * D_ + k0 + hlf * 16];
                a.p[1] = *(const v8bf*)&Ab[mlane * D_ + k0 + hlf * 16 + 8];
                v16bf bfr = *(const v16bf*)(Wo_h +
                        (size_t)(wave * 16 + mlane) * D_ + k0 + hlf * 16);
                c = wmma_bf16(a.v, bfr, c);
            }
            #pragma unroll
            for (int j = 0; j < 8; ++j) {
                int m = hlf * 8 + j;
                int d = wave * 16 + mlane;
                Xb[m * D_ + d] = (__bf16)c[j];
                if (step == STEPS_ - 1)
                    x_out[((size_t)(b0 + m) * H_ + h) * D_ + d] = c[j];
            }
        }
        __syncthreads();
    }
}

extern "C" void kernel_launch(void* const* d_in, const int* in_sizes, int n_in,
                              void* d_out, int out_size, void* d_ws, size_t ws_size,
                              hipStream_t stream) {
    (void)in_sizes; (void)n_in; (void)out_size; (void)ws_size;
    const float* x  = (const float*)d_in[0];
    const float* k  = (const float*)d_in[1];
    const float* v  = (const float*)d_in[2];
    const float* Wq = (const float*)d_in[3];
    const float* Wk = (const float*)d_in[4];
    const float* Wv = (const float*)d_in[5];
    const float* Wo = (const float*)d_in[6];

    float* out   = (float*)d_out;
    float* k_out = out;
    float* v_out = out + (size_t)B_ * H_ * S_ * D_;
    float* x_out = out + (size_t)2 * B_ * H_ * S_ * D_;
    __bf16* Wt   = (__bf16*)d_ws;   // 4 MB of bf16 transposed weights

    // one-time weight convert/transpose (cheap: 2M elements)
    int n = 4 * H_ * D_ * D_;
    cvt_weights<<<n / 256, 256, 0, stream>>>(Wq, Wk, Wv, Wo, Wt);

    size_t smem = (size_t)2 * TB * D_ * sizeof(__bf16)          // Xb + Ab
                + (size_t)(TB * D_ + TB * S_) * sizeof(float)   // Qs + Sc
                + (size_t)2 * TB * STEPS_ * D_ * sizeof(__bf16);// Kd + Vd
    (void)hipFuncSetAttribute((const void*)attn_decode,
                              hipFuncAttributeMaxDynamicSharedMemorySize,
                              (int)smem);
    attn_decode<<<dim3(H_ * (B_ / TB)), dim3(256), smem, stream>>>(
        x, k, v, Wt, k_out, v_out, x_out);
}
